// GCN_mamba_Net_Encoder_14422500180556
// MI455X (gfx1250) — matrix operand order repaired and checked by reading, not compiled, not run on hardware
//
#include <hip/hip_runtime.h>
#include <hip/hip_bf16.h>
#include <cstdint>

// ---------------------------------------------------------------------------
// GCN + Mamba encoder for MI455X (gfx1250, wave32, WMMA, async-LDS DMA).
// GEMMs with N%64==0 use an LDS-double-buffered pipeline: B panels streamed
// with global_load_async_to_lds_b128 (ASYNCcnt) and consumed by
// v_wmma_f32_16x16x32_bf16; the N=48 GEMM uses a direct per-wave WMMA kernel.
// Mamba scan is a chunked parallel linear-recurrence scan (3 passes).
// Workspace requirement: ~55M floats (~219 MB).
// ---------------------------------------------------------------------------

#define N_NODES 16384
#define F_IN 512
#define D_MODEL 256
#define N_EDGES 524288
#define D_INNER 512
#define D_STATE 16
#define D_CONV 4
#define DT_RANK 16
#define EPS 1e-5f

#define SCAN_CHUNKS 64
#define CHUNK_LEN 256                      // 64 * 256 = 16384
#define DN (D_INNER * D_STATE)             // 8192 independent recurrences

typedef __attribute__((ext_vector_type(16))) __bf16 v16bf;
typedef __attribute__((ext_vector_type(8)))  float  v8f;

__device__ __forceinline__ unsigned lds_addr_of(const void* p) {
  // AMDGPU flat pointers to LDS carry the LDS byte offset in the low 32 bits
  // (aperture check truncates to the low 32 bits, ISA 10.2).
  return (unsigned)(uintptr_t)p;
}

// ---------------------------------------------------------------------------
// GEMM (async-LDS pipeline): C[M x Nn] = A[M x K] @ B[K x Nn].
// Block = 256 threads = 8 waves computing a 32x64 macro-tile (2x4 WMMA tiles).
// Per K-step a 32x64 fp32 B panel is DMA'd to LDS with
// global_load_async_to_lds_b128 (double-buffered, s_wait_asynccnt fencing);
// A fragments load directly (row-contiguous pairs). Requires Nn%64==0,
// M%32==0, K%32==0; grid must be exactly (M/32)*(Nn/64).
// ---------------------------------------------------------------------------
__global__ void wmma_gemm_bf16_lds(const float* __restrict__ A,
                                   const float* __restrict__ B,
                                   float* __restrict__ C,
                                   int M, int Nn, int K) {
  __shared__ float bpanel[2][32 * 64];                 // 16 KB double buffer

  const int lane = threadIdx.x & 31;
  const int wave = threadIdx.x >> 5;
  const int nblk = Nn >> 6;
  const int m0 = (blockIdx.x / nblk) << 5;
  const int n0 = (blockIdx.x % nblk) << 6;

  const int tmL = wave >> 2;                           // 0..1
  const int tnL = wave & 3;                            // 0..3

  // A fragment addressing (16-bit A 16x32 layout, ISA 7.12.2)
  const int row_a   = m0 + (tmL << 4) + (lane & 15);
  const int kbase_a = (lane >> 4) << 3;                // 0 or 8
  // B fragment addressing out of the LDS panel
  const int ncol    = (tnL << 4) + (lane & 15);        // 0..63
  const int kbase_b = (lane >> 4) << 3;                // 0 or 8

  // cooperative B-panel copy assignment: 256 thr x 8 floats = 32x64 panel
  const int krow = threadIdx.x >> 3;                   // 0..31
  const int seg  = (threadIdx.x & 7) << 3;             // 0,8,...,56

  // prologue: stage panel for kk = 0 into buffer 0
  {
    const unsigned ldsa = lds_addr_of(&bpanel[0][krow * 64 + seg]);
    const float* gsrc = B + (size_t)krow * Nn + n0 + seg;
    asm volatile("global_load_async_to_lds_b128 %0, %1, off"
                 :: "v"(ldsa), "v"(gsrc) : "memory");
    asm volatile("global_load_async_to_lds_b128 %0, %1, off offset:16"
                 :: "v"(ldsa), "v"(gsrc) : "memory");
  }

  v8f acc = {};
  int cur = 0;
  for (int kk = 0; kk < K; kk += 32) {
    asm volatile("s_wait_asynccnt 0x0" ::: "memory");  // own DMA done
    __syncthreads();                                   // everyone's DMA visible

    if (kk + 32 < K) {                                 // prefetch next panel
      const unsigned ldsa = lds_addr_of(&bpanel[cur ^ 1][krow * 64 + seg]);
      const float* gsrc = B + (size_t)(kk + 32 + krow) * Nn + n0 + seg;
      asm volatile("global_load_async_to_lds_b128 %0, %1, off"
                   :: "v"(ldsa), "v"(gsrc) : "memory");
      asm volatile("global_load_async_to_lds_b128 %0, %1, off offset:16"
                   :: "v"(ldsa), "v"(gsrc) : "memory");
    }

    v16bf a, b;
    const float* ap = A + (size_t)row_a * K + kk;
#pragma unroll
    for (int j = 0; j < 8; ++j) {
      int k0 = ((j >> 2) << 4) + kbase_a + ((j & 3) << 1);
      a[2 * j]     = (__bf16)ap[k0];
      a[2 * j + 1] = (__bf16)ap[k0 + 1];
    }
    const float* bp = &bpanel[cur][0];
#pragma unroll
    for (int j = 0; j < 8; ++j) {
      int k = ((j >> 2) << 4) + kbase_b + ((j & 3) << 1);
      b[2 * j]     = (__bf16)bp[k * 64 + ncol];        // ds_load_b32
      b[2 * j + 1] = (__bf16)bp[(k + 1) * 64 + ncol];
    }
    acc = __builtin_amdgcn_wmma_f32_16x16x32_bf16(
        false, a, false, b, (short)0, acc, false, false);

    __syncthreads();                                   // reads of cur done
    cur ^= 1;
  }

  // D layout: VGPR i -> row i (lanes 0-15) / row i+8 (lanes 16-31)
  const int rbase = m0 + (tmL << 4) + ((lane >> 4) << 3);
  const int c = n0 + ncol;
#pragma unroll
  for (int i = 0; i < 8; ++i)
    C[(size_t)(rbase + i) * Nn + c] = acc[i];
}

// ---------------------------------------------------------------------------
// Direct GEMM (for Nn=48 x_proj): one wave per 16x16 C tile, 8 waves/block.
// ---------------------------------------------------------------------------
__global__ void wmma_gemm_bf16(const float* __restrict__ A,
                               const float* __restrict__ B,
                               float* __restrict__ C,
                               int M, int Nn, int K) {
  const int lane = threadIdx.x & 31;
  const int wave = threadIdx.x >> 5;
  const int ntn  = Nn >> 4;
  const int tile = blockIdx.x * 8 + wave;
  const int total = (M >> 4) * ntn;
  if (tile >= total) return;                 // uniform across the wave
  const int tm = tile / ntn;
  const int tn = tile - tm * ntn;

  const int row_a   = (tm << 4) + (lane & 15);
  const int kbase_a = (lane >> 4) << 3;
  const int col_b   = (tn << 4) + (lane & 15);
  const int kbase_b = (lane >> 4) << 3;

  v8f acc = {};
  for (int kk = 0; kk < K; kk += 32) {
    const float* ap = A + (size_t)row_a * K + kk;
    __builtin_prefetch(ap + 32, 0, 0);
    v16bf a, b;
#pragma unroll
    for (int j = 0; j < 8; ++j) {
      int k0 = ((j >> 2) << 4) + kbase_a + ((j & 3) << 1);
      a[2 * j]     = (__bf16)ap[k0];
      a[2 * j + 1] = (__bf16)ap[k0 + 1];
    }
    const float* bp = B + col_b;
#pragma unroll
    for (int j = 0; j < 8; ++j) {
      int k = ((j >> 2) << 4) + kbase_b + ((j & 3) << 1);
      b[2 * j]     = (__bf16)bp[(size_t)(kk + k) * Nn];
      b[2 * j + 1] = (__bf16)bp[(size_t)(kk + k + 1) * Nn];
    }
    acc = __builtin_amdgcn_wmma_f32_16x16x32_bf16(
        false, a, false, b, (short)0, acc, false, false);
  }
  const int rbase = (tm << 4) + ((lane >> 4) << 3);
  const int c = (tn << 4) + (lane & 15);
#pragma unroll
  for (int i = 0; i < 8; ++i)
    C[(size_t)(rbase + i) * Nn + c] = acc[i];
}

// --------------------------- batchnorm helpers -----------------------------
__global__ void stats_kernel(const float* __restrict__ x, float* __restrict__ stats) {
  const int col = threadIdx.x;
  const int r0 = blockIdx.x * 64;
  float s = 0.f, q = 0.f;
  for (int r = 0; r < 64; ++r) {
    float v = x[(size_t)(r0 + r) * D_MODEL + col];
    s += v; q += v * v;
  }
  atomicAdd(&stats[col], s);
  atomicAdd(&stats[D_MODEL + col], q);
}

__global__ void add_stats_kernel(const float* __restrict__ a,
                                 const float* __restrict__ b,
                                 float* __restrict__ s_out,
                                 float* __restrict__ stats) {
  const int col = threadIdx.x;
  const int r0 = blockIdx.x * 64;
  float s = 0.f, q = 0.f;
  for (int r = 0; r < 64; ++r) {
    size_t idx = (size_t)(r0 + r) * D_MODEL + col;
    float v = a[idx] + b[idx];
    s_out[idx] = v;
    s += v; q += v * v;
  }
  atomicAdd(&stats[col], s);
  atomicAdd(&stats[D_MODEL + col], q);
}

__global__ void bn_apply_kernel(const float* __restrict__ x,
                                const float* __restrict__ stats,
                                const float* __restrict__ g,
                                const float* __restrict__ bb,
                                float* __restrict__ out,
                                int relu, int rows) {
  const int col = threadIdx.x;
  const size_t idx = (size_t)blockIdx.x * D_MODEL + col;
  const float inv_n = 1.f / (float)rows;
  const float mean = stats[col] * inv_n;
  const float var = stats[D_MODEL + col] * inv_n - mean * mean;
  float v = (x[idx] - mean) * rsqrtf(var + EPS) * g[col] + bb[col];
  if (relu) v = fmaxf(v, 0.f);
  out[idx] = v;
}

// ------------------------- GCN edge scatter (segment_sum) ------------------
__global__ void edge_kernel(const int* __restrict__ er, const int* __restrict__ ec,
                            const float* __restrict__ ew,
                            const float* __restrict__ support,
                            float* __restrict__ xg) {
  const int e = blockIdx.x;
  const int row = er[e];
  const int col = ec[e];
  const float w = ew[e];
  const int f = threadIdx.x;
  atomicAdd(&xg[(size_t)row * D_MODEL + f], w * support[(size_t)col * D_MODEL + f]);
}

// ---------------------- causal depthwise conv + SiLU -----------------------
__global__ void conv_silu_kernel(const float* __restrict__ xr,
                                 const float* __restrict__ cw,
                                 const float* __restrict__ cb,
                                 float* __restrict__ xc) {
  const int idx = blockIdx.x * 256 + threadIdx.x;     // N*512 elements
  const int l = idx >> 9;
  const int d = idx & 511;
  float acc = cb[d];
#pragma unroll
  for (int k = 0; k < D_CONV; ++k) {
    int t = l - (D_CONV - 1) + k;
    if (t >= 0) acc += xr[(size_t)t * (2 * D_INNER) + d] * cw[d * D_CONV + k];
  }
  xc[idx] = acc / (1.f + __expf(-acc));
}

// ----------------- delta = softplus(dr @ dt_proj_w + dt_proj_b) ------------
__global__ void delta_kernel(const float* __restrict__ dbc,
                             const float* __restrict__ w,
                             const float* __restrict__ bias,
                             float* __restrict__ delta) {
  const int idx = blockIdx.x * 256 + threadIdx.x;     // N*512
  const int l = idx >> 9;
  const int d = idx & 511;
  float acc = bias[d];
#pragma unroll
  for (int r = 0; r < DT_RANK; ++r)
    acc += dbc[(size_t)l * 48 + r] * w[r * D_INNER + d];
  delta[idx] = (acc > 20.f) ? acc : __logf(1.f + __expf(acc));
}

// ------------------------- chunked parallel SSM scan -----------------------
__global__ void scan_passA(const float* __restrict__ delta,
                           const float* __restrict__ xc,
                           const float* __restrict__ dbc,
                           const float* __restrict__ A_log,
                           float* __restrict__ P, float* __restrict__ H) {
  const int n = threadIdx.x & 15;
  const int d = blockIdx.y * 16 + (threadIdx.x >> 4);
  const int chunk = blockIdx.x;
  const float A = -__expf(A_log[d * D_STATE + n]);
  float p = 1.f, h = 0.f;
  int l = chunk * CHUNK_LEN;
  for (int i = 0; i < CHUNK_LEN; ++i, ++l) {
    const float dv = delta[(size_t)l * D_INNER + d];
    const float dA = __expf(dv * A);
    h = dA * h + dv * dbc[(size_t)l * 48 + DT_RANK + n] * xc[(size_t)l * D_INNER + d];
    p *= dA;
  }
  const int idx = chunk * DN + d * D_STATE + n;
  P[idx] = p;
  H[idx] = h;
}

__global__ void scan_mid_kernel(const float* __restrict__ P,
                                const float* __restrict__ H,
                                float* __restrict__ I) {
  const int j = blockIdx.x * 256 + threadIdx.x;       // 8192 recurrences
  float h = 0.f;
  for (int c = 0; c < SCAN_CHUNKS; ++c) {
    I[c * DN + j] = h;
    h = P[c * DN + j] * h + H[c * DN + j];
  }
}

__global__ void scan_passC(const float* __restrict__ delta,
                           const float* __restrict__ xc,
                           const float* __restrict__ dbc,
                           const float* __restrict__ xr,
                           const float* __restrict__ A_log,
                           const float* __restrict__ d_skip,
                           const float* __restrict__ I,
                           float* __restrict__ y) {
  const int n = threadIdx.x & 15;
  const int d = blockIdx.y * 16 + (threadIdx.x >> 4);
  const int chunk = blockIdx.x;
  const float A = -__expf(A_log[d * D_STATE + n]);
  const float skip = d_skip[d];
  float h = I[chunk * DN + d * D_STATE + n];
  int l = chunk * CHUNK_LEN;
  for (int i = 0; i < CHUNK_LEN; ++i, ++l) {
    const float dv = delta[(size_t)l * D_INNER + d];
    const float xcv = xc[(size_t)l * D_INNER + d];
    const float dA = __expf(dv * A);
    h = dA * h + dv * dbc[(size_t)l * 48 + DT_RANK + n] * xcv;
    float contrib = h * dbc[(size_t)l * 48 + DT_RANK + D_STATE + n];
    contrib += __shfl_xor(contrib, 1, 16);
    contrib += __shfl_xor(contrib, 2, 16);
    contrib += __shfl_xor(contrib, 4, 16);
    contrib += __shfl_xor(contrib, 8, 16);
    if (n == 0) {
      const float res = xr[(size_t)l * (2 * D_INNER) + D_INNER + d];
      y[(size_t)l * D_INNER + d] =
          (contrib + xcv * skip) * (res / (1.f + __expf(-res)));
    }
  }
}

// ---------------------------------------------------------------------------
extern "C" void kernel_launch(void* const* d_in, const int* in_sizes, int n_in,
                              void* d_out, int out_size, void* d_ws, size_t ws_size,
                              hipStream_t stream) {
  const float* x         = (const float*)d_in[0];
  const int*   er        = (const int*)  d_in[1];
  const int*   ec        = (const int*)  d_in[2];
  const float* ew        = (const float*)d_in[3];
  const float* w_emb     = (const float*)d_in[4];
  const float* bn_in_g   = (const float*)d_in[5];
  const float* bn_in_b   = (const float*)d_in[6];
  const float* gcn_w     = (const float*)d_in[7];
  const float* in_proj_w = (const float*)d_in[8];
  const float* conv_w    = (const float*)d_in[9];
  const float* conv_b    = (const float*)d_in[10];
  const float* x_proj_w  = (const float*)d_in[11];
  const float* dt_proj_w = (const float*)d_in[12];
  const float* dt_proj_b = (const float*)d_in[13];
  const float* A_log     = (const float*)d_in[14];
  const float* d_skip    = (const float*)d_in[15];
  const float* out_proj_w= (const float*)d_in[16];
  const float* bn_loc_g  = (const float*)d_in[17];
  const float* bn_loc_b  = (const float*)d_in[18];
  float* out = (float*)d_out;

  // workspace carve-up (floats); ~55M floats total (~219 MB)
  float* ws = (float*)d_ws;
  const size_t NM = (size_t)N_NODES * D_MODEL;          // 4M
  float* y_emb   = ws;                                   // x_emb (in place); later pre-bn sum
  float* support = ws + NM;                              // later m_out
  float* x_gcn   = ws + 2 * NM;
  float* xr      = ws + 3 * NM;                          // N x 1024
  float* xc      = xr + (size_t)N_NODES * 2 * D_INNER;   // N x 512
  float* dbc     = xc + (size_t)N_NODES * D_INNER;       // N x 48
  float* delta   = dbc + (size_t)N_NODES * 48;           // N x 512
  float* yb      = delta + (size_t)N_NODES * D_INNER;    // N x 512
  float* stats   = yb + (size_t)N_NODES * D_INNER;       // 512
  float* chunkP  = stats + 1024;
  float* chunkH  = chunkP + (size_t)SCAN_CHUNKS * DN;
  float* chunkI  = chunkH + (size_t)SCAN_CHUNKS * DN;

  // 1) x @ w_emb  (16384x512 @ 512x256): (M/32)*(N/64) = 512*4 blocks
  wmma_gemm_bf16_lds<<<2048, 256, 0, stream>>>(x, w_emb, y_emb, N_NODES, D_MODEL, F_IN);

  // 2) batchnorm + relu (in place -> x_emb)
  hipMemsetAsync(stats, 0, 1024 * sizeof(float), stream);
  stats_kernel<<<N_NODES / 64, D_MODEL, 0, stream>>>(y_emb, stats);
  bn_apply_kernel<<<N_NODES, D_MODEL, 0, stream>>>(y_emb, stats, bn_in_g, bn_in_b,
                                                   y_emb, 1, N_NODES);

  // 3) support = x_emb @ gcn_w  (K=256)
  wmma_gemm_bf16_lds<<<2048, 256, 0, stream>>>(y_emb, gcn_w, support, N_NODES, D_MODEL, D_MODEL);

  // 4) segment-sum of weighted messages
  hipMemsetAsync(x_gcn, 0, NM * sizeof(float), stream);
  edge_kernel<<<N_EDGES, D_MODEL, 0, stream>>>(er, ec, ew, support, x_gcn);

  // 5) xr = x_gcn @ in_proj_w  (16384x256 @ 256x1024): 512*16 blocks
  wmma_gemm_bf16_lds<<<8192, 256, 0, stream>>>(x_gcn, in_proj_w, xr, N_NODES, 2 * D_INNER, D_MODEL);

  // 6) causal depthwise conv + SiLU -> xc
  conv_silu_kernel<<<(N_NODES * D_INNER) / 256, 256, 0, stream>>>(xr, conv_w, conv_b, xc);

  // 7) dbc = xc @ x_proj_w  (16384x512 @ 512x48): N=48 -> direct kernel
  wmma_gemm_bf16<<<384, 256, 0, stream>>>(xc, x_proj_w, dbc, N_NODES, 48, D_INNER);

  // 8) delta = softplus(dr @ dt_proj_w + b)  (K=16, VALU)
  delta_kernel<<<(N_NODES * D_INNER) / 256, 256, 0, stream>>>(dbc, dt_proj_w, dt_proj_b, delta);

  // 9) chunked parallel scan -> y (fused +xc*d_skip and *silu(res))
  scan_passA<<<dim3(SCAN_CHUNKS, D_INNER / 16), 256, 0, stream>>>(delta, xc, dbc, A_log,
                                                                  chunkP, chunkH);
  scan_mid_kernel<<<DN / 256, 256, 0, stream>>>(chunkP, chunkH, chunkI);
  scan_passC<<<dim3(SCAN_CHUNKS, D_INNER / 16), 256, 0, stream>>>(delta, xc, dbc, xr, A_log,
                                                                  d_skip, chunkI, yb);

  // 10) m_out = y @ out_proj_w  (16384x512 @ 512x256) -> reuse `support`
  wmma_gemm_bf16_lds<<<2048, 256, 0, stream>>>(yb, out_proj_w, support, N_NODES, D_MODEL, D_INNER);

  // 11) final batchnorm of (x_gcn + m_out)
  hipMemsetAsync(stats, 0, 1024 * sizeof(float), stream);
  add_stats_kernel<<<N_NODES / 64, D_MODEL, 0, stream>>>(x_gcn, support, y_emb, stats);
  bn_apply_kernel<<<N_NODES, D_MODEL, 0, stream>>>(y_emb, stats, bn_loc_g, bn_loc_b,
                                                   out, 0, N_NODES);
}
